// DiscriminativeLoss_69569880261306
// MI455X (gfx1250) — compile-verified, matrix operand order checked by reading, not run on hardware
//
#include <hip/hip_runtime.h>
#include <hip/hip_bf16.h>

// Discriminative loss, MI455X (gfx1250, wave32).
// Pass 1: segment-sum as one-hot GEMM via v_wmma_f32_16x16x32_f16
//         (instances 1..16 = exactly one 16-wide WMMA tile; background never needed).
// Pass 2: streaming per-pixel hinge with LDS-cached means (stride-17 padded).
// Both passes are HBM-bound: ~272MB total -> ~12us at 23.3 TB/s.

typedef _Float16 v16h __attribute__((ext_vector_type(16)));
typedef float    v8f  __attribute__((ext_vector_type(8)));

#define Bn 8
#define Cn 16
#define Nn (512 * 512)

// ws layout (floats):
//   sums   [B][16][16] @ 0      (instance k stored at k-1)
//   counts [B][16]     @ 2048
//   means  [B][16][16] @ 2176
//   hinge  [B][16]     @ 4224
#define WS_SUMS   0
#define WS_CNT    2048
#define WS_MEANS  2176
#define WS_HINGE  4224
#define WS_TOTAL  4352

#define WAVES_PER_BATCH 256
#define PIX_PER_WAVE    (Nn / WAVES_PER_BATCH)   // 1024
#define P2_BLOCKS_PER_B 64
#define P2_PIX_PER_BLK  (Nn / P2_BLOCKS_PER_B)   // 4096

__global__ __launch_bounds__(256) void dl_init(float* __restrict__ ws) {
    for (int i = threadIdx.x; i < WS_TOTAL; i += 256) ws[i] = 0.0f;
}

// ---------------- Pass 1: sums & counts via one-hot WMMA ----------------
// Wave processes PIX_PER_WAVE contiguous pixels of one batch in 32-pixel chunks.
// A (16x32 f16) = onehot[inst-1][pixel], B (32x16 f16) = emb[pixel][channel],
// C/D (16x16 f32) accumulates sums[inst-1][channel] in registers.
__global__ __launch_bounds__(256) void dl_pass1(const float* __restrict__ emb,
                                                const int* __restrict__ lab,
                                                float* __restrict__ ws) {
    float* sums   = ws + WS_SUMS;
    float* counts = ws + WS_CNT;

    const int wave = (blockIdx.x * blockDim.x + threadIdx.x) >> 5;
    const int lane = threadIdx.x & 31;
    const int b      = wave / WAVES_PER_BATCH;
    const int pstart = (wave % WAVES_PER_BATCH) * PIX_PER_WAVE;

    const int ch     = lane & 15;             // A row M, B col N owned by this lane
    const int off    = (lane < 16) ? 0 : 8;   // K-half per ISA 16-bit layout
    const int m_inst = ch + 1;                // instance id this lane's A-row encodes

    const float* embB = emb + ((size_t)b * Cn + ch) * (size_t)Nn;
    const int*   labB = lab + (size_t)b * Nn;

    v8f   acc = {};
    float cnt = 0.0f;

    for (int p0 = pstart; p0 < pstart + PIX_PER_WAVE; p0 += 32) {
        // Labels this lane needs: K = off+{0..7} and off+16+{0..7}
        const int* lp = labB + p0 + off;
        const int4 L0 = *(const int4*)(lp);
        const int4 L1 = *(const int4*)(lp + 4);
        const int4 L2 = *(const int4*)(lp + 16);
        const int4 L3 = *(const int4*)(lp + 20);
        // Embedding (channel ch) at the same pixel indices
        const float* ep = embB + p0 + off;
        const float4 E0 = *(const float4*)(ep);
        const float4 E1 = *(const float4*)(ep + 4);
        const float4 E2 = *(const float4*)(ep + 16);
        const float4 E3 = *(const float4*)(ep + 20);

        int labs[16] = {L0.x, L0.y, L0.z, L0.w, L1.x, L1.y, L1.z, L1.w,
                        L2.x, L2.y, L2.z, L2.w, L3.x, L3.y, L3.z, L3.w};
        float es[16] = {E0.x, E0.y, E0.z, E0.w, E1.x, E1.y, E1.z, E1.w,
                        E2.x, E2.y, E2.z, E2.w, E3.x, E3.y, E3.z, E3.w};

        v16h a, bb;
        int mc = 0;
#pragma unroll
        for (int j = 0; j < 16; ++j) {
            const bool hit = (labs[j] == m_inst);
            a[j]  = hit ? (_Float16)1.0f : (_Float16)0.0f;
            mc   += hit ? 1 : 0;
            bb[j] = (_Float16)es[j];
        }
        cnt += (float)mc;

        acc = __builtin_amdgcn_wmma_f32_16x16x32_f16(
            /*neg_a=*/false, a, /*neg_b=*/false, bb,
            /*c_mod=*/(short)0, acc, /*reuse_a=*/false, /*reuse_b=*/false);
    }

    // D layout: lane L, VGPR j -> M = j + (L<16?0:8), N = L%16
#pragma unroll
    for (int j = 0; j < 8; ++j) {
        const int m = j + ((lane < 16) ? 0 : 8);   // instance m+1 at slot m
        atomicAdd(&sums[(b * 16 + m) * 16 + ch], acc[j]);
    }
    // lane L and L+16 jointly cover all 32 pixels of each chunk for instance ch+1
    atomicAdd(&counts[b * 16 + ch], cnt);
}

// ---------------- means = sums / max(counts, 1) ----------------
__global__ __launch_bounds__(256) void dl_means(float* __restrict__ ws) {
    const int b = blockIdx.x;
    const int t = threadIdx.x;        // t = k*16 + c, k in 0..15 (inst k+1)
    const int k = t >> 4;
    const float c = fmaxf(ws[WS_CNT + b * 16 + k], 1.0f);
    ws[WS_MEANS + b * 256 + t] = ws[WS_SUMS + b * 256 + t] / c;
}

// ---------------- Pass 2: per-pixel variance hinge ----------------
__global__ __launch_bounds__(256) void dl_pass2(const float* __restrict__ emb,
                                                const int* __restrict__ lab,
                                                float* __restrict__ ws) {
    __shared__ float smean[16 * 17];   // stride 17: no bank aliasing on label gather
    __shared__ float shinge[17];

    const int tid   = threadIdx.x;
    const int b     = blockIdx.x / P2_BLOCKS_PER_B;
    const int pbase = (blockIdx.x % P2_BLOCKS_PER_B) * P2_PIX_PER_BLK;

    if (tid < 17) shinge[tid] = 0.0f;
    {   // means[b][k][c] -> smean[k*17+c]
        const int k = tid >> 4, c = tid & 15;
        smean[k * 17 + c] = ws[WS_MEANS + b * 256 + tid];
    }
    __syncthreads();

    const float* embB = emb + (size_t)b * Cn * (size_t)Nn;
    const int*   labB = lab + (size_t)b * Nn;

    for (int i = 0; i < P2_PIX_PER_BLK; i += 256) {
        const int p = pbase + i + tid;
        const int l = labB[p];
        if (l != 0) {
            const float* mrow = &smean[(l - 1) * 17];
            const float* ep   = embB + p;
            float sqd = 0.0f;
#pragma unroll
            for (int c = 0; c < 16; ++c) {
                const float dv = ep[(size_t)c * Nn] - mrow[c];
                sqd += dv * dv;
            }
            const float u = sqrtf(sqd) - 0.5f;     // DELTA_V = 0.5
            if (u > 0.0f) atomicAdd(&shinge[l], u * u);
        }
    }
    __syncthreads();
    if (tid >= 1 && tid <= 16)
        atomicAdd(&ws[WS_HINGE + b * 16 + (tid - 1)], shinge[tid]);
}

// ---------------- Final: distance/reg terms + batch averaging ----------------
__global__ __launch_bounds__(64) void dl_final(const float* __restrict__ ws,
                                               float* __restrict__ out) {
    __shared__ float s_var[8], s_dist[8], s_reg[8], s_valid[8];
    const int t = threadIdx.x;
    if (t < 8) {
        const float* cntp = ws + WS_CNT + t * 16;
        const float* hng  = ws + WS_HINGE + t * 16;
        const float* mn   = ws + WS_MEANS + t * 256;

        float n_inst = 0.0f;
        float var = 0.0f, reg = 0.0f;
        for (int k = 0; k < 16; ++k) {
            if (cntp[k] > 0.0f) {
                n_inst += 1.0f;
                var += hng[k] / fmaxf(cntp[k], 1.0f);
                float nm = 0.0f;
                for (int c = 0; c < 16; ++c) nm += mn[k * 16 + c] * mn[k * 16 + c];
                reg += sqrtf(nm);
            }
        }
        const float denom = fmaxf(n_inst, 1.0f);
        var /= denom;
        reg /= denom;

        float hs = 0.0f;
        for (int i = 0; i < 16; ++i) {
            if (!(cntp[i] > 0.0f)) continue;
            for (int j = i + 1; j < 16; ++j) {
                if (!(cntp[j] > 0.0f)) continue;
                float sq = 0.0f;
                for (int c = 0; c < 16; ++c) {
                    const float dv = mn[i * 16 + c] - mn[j * 16 + c];
                    sq += dv * dv;
                }
                const float u = 3.0f - sqrtf(sq);  // 2*DELTA_D = 3.0
                if (u > 0.0f) hs += u * u;
            }
        }
        const float npairs = n_inst * (n_inst - 1.0f) * 0.5f;
        const float dist = (n_inst > 1.0f) ? hs / fmaxf(npairs, 1.0f) : 0.0f;

        s_var[t] = var;
        s_dist[t] = dist;
        s_reg[t] = reg;
        s_valid[t] = (n_inst > 0.0f) ? 1.0f : 0.0f;
    }
    __syncthreads();
    if (t == 0) {
        float nv = 0.0f, tv = 0.0f, td = 0.0f, tr = 0.0f;
        for (int b = 0; b < 8; ++b) {
            nv += s_valid[b];
            tv += s_var[b] * s_valid[b];
            td += s_dist[b] * s_valid[b];
            tr += s_reg[b] * s_valid[b];
        }
        nv = fmaxf(nv, 1.0f);
        tv /= nv; td /= nv; tr /= nv;
        out[0] = tv + td + 0.001f * tr;   // ALPHA=BETA=1, GAMMA=0.001
        out[1] = tv;
        out[2] = td;
        out[3] = tr;
    }
}

extern "C" void kernel_launch(void* const* d_in, const int* in_sizes, int n_in,
                              void* d_out, int out_size, void* d_ws, size_t ws_size,
                              hipStream_t stream) {
    const float* emb = (const float*)d_in[0];
    const int*   lab = (const int*)d_in[1];
    float*       out = (float*)d_out;
    float*       ws  = (float*)d_ws;

    dl_init <<<1, 256, 0, stream>>>(ws);
    // 2M pixels / 1024 pixels-per-wave = 2048 waves = 256 blocks x 8 waves
    dl_pass1<<<dim3((Bn * WAVES_PER_BATCH) / 8), dim3(256), 0, stream>>>(emb, lab, ws);
    dl_means<<<dim3(Bn), dim3(256), 0, stream>>>(ws);
    dl_pass2<<<dim3(Bn * P2_BLOCKS_PER_B), dim3(256), 0, stream>>>(emb, lab, ws);
    dl_final<<<1, 64, 0, stream>>>(ws, out);
}
